// Attention_72052371358090
// MI455X (gfx1250) — matrix operand order repaired and checked by reading, not compiled, not run on hardware
//
#include <hip/hip_runtime.h>
#include <hip/hip_bf16.h>

// ---------------- problem constants ----------------
#define BATCH   2
#define SEQ     2048
#define DIM     4096
#define NH      32
#define NKVH    8
#define HD      128
#define MTOT    (BATCH * SEQ)          // 4096 rows of x
#define QCOLS   (NH * HD)              // 4096
#define KVCOLS  (NKVH * HD)            // 1024
#define GQA     (NH / NKVH)            // 4

typedef __bf16 bf16_t;
typedef __attribute__((ext_vector_type(16))) __bf16 v16bf;
typedef __attribute__((ext_vector_type(8)))  float  v8f;

// ---------------- small helpers ----------------
__device__ inline v8f v8f_zero() {
    v8f z;
#pragma unroll
    for (int i = 0; i < 8; ++i) z[i] = 0.0f;
    return z;
}

__device__ inline v8f wmma_bf16(v16bf a, v16bf b, v8f c) {
    // D = A(16x32 bf16) * B(32x16 bf16) + C(16x16 f32)
    return __builtin_amdgcn_wmma_f32_16x16x32_bf16(
        /*neg_a=*/false, a, /*neg_b=*/false, b,
        /*c_mod=*/(short)0, c, /*reuse_a=*/false, /*reuse_b=*/false);
}

// load a v16bf fragment from two 16-byte chunks (A-matrix layout: two
// 8-element runs per lane; B-matrix layout: same two runs, p1 = p0+16)
__device__ inline v16bf ldfrag2(const bf16_t* p0, const bf16_t* p1) {
    union { uint4 u[2]; v16bf v; } r;
    r.u[0] = *reinterpret_cast<const uint4*>(p0);
    r.u[1] = *reinterpret_cast<const uint4*>(p1);
    return r.v;
}

// ---- CDNA5 async global->LDS copy (per-lane 16B), tracked by ASYNCcnt ----
__device__ inline uint32_t lds_off32(const void* p) {
    // low 32 bits of a flat LDS-aperture address are the DS address (ISA 10.2)
    return (uint32_t)(uintptr_t)p;
}
__device__ inline void async_b128(uint32_t lds_dst, const void* gsrc) {
    asm volatile("global_load_async_to_lds_b128 %0, %1, off"
                 :: "v"(lds_dst), "v"((unsigned long long)(uintptr_t)gsrc)
                 : "memory");
}
__device__ inline void wait_async0() {
    asm volatile("s_wait_asynccnt 0x0" ::: "memory");
}

// ---------------- fp32 -> bf16 convert ----------------
__global__ __launch_bounds__(256) void cvt_bf16_kernel(const float* __restrict__ src,
                                                       bf16_t* __restrict__ dst,
                                                       long n) {
    long i = (long)blockIdx.x * blockDim.x + threadIdx.x;
    long stride = (long)gridDim.x * blockDim.x;
    for (; i < n; i += stride) dst[i] = (bf16_t)src[i];
}

// ---------------- RoPE in-place on bf16 [rows, heads*128] ----------------
__global__ __launch_bounds__(256) void rope_kernel(bf16_t* __restrict__ x,
                                                   const float* __restrict__ cs,
                                                   const float* __restrict__ sn,
                                                   int heads) {
    long total = (long)MTOT * heads * (HD / 2);
    long i = (long)blockIdx.x * blockDim.x + threadIdx.x;
    long stride = (long)gridDim.x * blockDim.x;
    for (; i < total; i += stride) {
        int  p   = (int)(i % (HD / 2));
        long t   = i / (HD / 2);
        int  hh  = (int)(t % heads);
        long row = t / heads;
        int  pos = (int)(row % SEQ);
        long base = row * (long)(heads * HD) + (long)hh * HD + (long)p * 2;
        float x0 = (float)x[base];
        float x1 = (float)x[base + 1];
        float c = cs[pos * (HD / 2) + p];
        float s = sn[pos * (HD / 2) + p];
        x[base]     = (bf16_t)(x0 * c - x1 * s);
        x[base + 1] = (bf16_t)(x0 * s + x1 * c);
    }
}

// ---------------- GEMM: C[M,N] = A[M,K] * W[N,K]^T (bf16 in, f32 acc) -----
// block tile 128x128, K-step 32, 256 threads = 8 waves, wave tile 64x32.
// Double-buffered LDS staged via async global->LDS; one barrier per K-step.
template <bool OUTF32>
__global__ __launch_bounds__(256) void gemm_bf16_nt(const bf16_t* __restrict__ A,
                                                    const bf16_t* __restrict__ W,
                                                    float* __restrict__ Cf,
                                                    bf16_t* __restrict__ Cb,
                                                    int M, int N, int K) {
    __shared__ __align__(16) bf16_t lA[2][128 * 32];
    __shared__ __align__(16) bf16_t lW[2][128 * 32];

    const int tid  = threadIdx.x;
    const int wave = tid >> 5;
    const int lane = tid & 31;
    const int l16  = lane & 15;
    const int lh   = lane >> 4;          // lane-half (wave32)
    const int wm   = wave & 1;           // 2 wave rows  -> 64 M each
    const int wn   = wave >> 1;          // 4 wave cols  -> 32 N each
    const int m0   = blockIdx.y * 128;
    const int n0   = blockIdx.x * 128;

    v8f acc[4][2];
#pragma unroll
    for (int i = 0; i < 4; ++i)
#pragma unroll
        for (int j = 0; j < 2; ++j) acc[i][j] = v8f_zero();

    const int ldrow = tid >> 1;          // 0..127
    const int ldko  = (tid & 1) * 16;    // 0 or 16

    // issue async copies for K-tile `kt` into buffer `buf` (32B per thread/tile)
    auto issue_tile = [&](int buf, int kt) {
        const bf16_t* ga = A + (size_t)(m0 + ldrow) * K + kt + ldko;
        const bf16_t* gw = W + (size_t)(n0 + ldrow) * K + kt + ldko;
        uint32_t sa = lds_off32(lA[buf] + ldrow * 32 + ldko);
        uint32_t sw = lds_off32(lW[buf] + ldrow * 32 + ldko);
        async_b128(sa,      ga);
        async_b128(sa + 16, ga + 8);
        async_b128(sw,      gw);
        async_b128(sw + 16, gw + 8);
    };

    issue_tile(0, 0);
    int buf = 0;

    for (int kt = 0; kt < K; kt += 32) {
        wait_async0();        // this wave's copies into `buf` have landed
        __syncthreads();      // everyone's copies landed; prev compute on buf^1 done
        if (kt + 32 < K) issue_tile(buf ^ 1, kt + 32);   // overlap next tile

        // ---- fragments ----
        v16bf af[4];
#pragma unroll
        for (int i = 0; i < 4; ++i) {
            const bf16_t* p = lA[buf] + (wm * 64 + i * 16 + l16) * 32 + lh * 8;
            af[i] = ldfrag2(p, p + 16);          // A layout: K runs {kh*8.., 16+kh*8..}
        }
        v16bf bfg[2];
#pragma unroll
        for (int j = 0; j < 2; ++j) {
            const bf16_t* p = lW[buf] + (wn * 32 + j * 16 + l16) * 32 + lh * 8;
            bfg[j] = ldfrag2(p, p + 16);         // B layout: K runs {kh*8.., 16+kh*8..}
        }
#pragma unroll
        for (int i = 0; i < 4; ++i)
#pragma unroll
            for (int j = 0; j < 2; ++j)
                acc[i][j] = wmma_bf16(af[i], bfg[j], acc[i][j]);
        buf ^= 1;
    }

    // ---- epilogue ----
#pragma unroll
    for (int i = 0; i < 4; ++i)
#pragma unroll
        for (int j = 0; j < 2; ++j)
#pragma unroll
            for (int r = 0; r < 8; ++r) {
                int gr = m0 + wm * 64 + i * 16 + r + 8 * lh;
                int gc = n0 + wn * 32 + j * 16 + l16;
                if (OUTF32) Cf[(size_t)gr * N + gc] = acc[i][j][r];
                else        Cb[(size_t)gr * N + gc] = (bf16_t)acc[i][j][r];
            }
}

// ---------------- flash attention (causal, GQA) ----------------
// grid: (SEQ/64, BATCH*NH). block: 128 threads = 4 waves, 16 query rows/wave.
__global__ __launch_bounds__(128) void flash_attn_kernel(const bf16_t* __restrict__ q,
                                                         const bf16_t* __restrict__ k,
                                                         const bf16_t* __restrict__ v,
                                                         bf16_t* __restrict__ o) {
    __shared__ __align__(16) bf16_t lQ[64 * HD];        // 16 KB
    __shared__ __align__(16) bf16_t lK[64 * HD];        // 16 KB, row-major [key][d]
    __shared__ __align__(16) bf16_t lV[HD * 64];        // 16 KB, transposed [d][key]
    __shared__ __align__(16) bf16_t lP[4][16 * 64];     // 8 KB per-wave P staging

    const int qt   = blockIdx.x;
    const int bh   = blockIdx.y;
    const int b    = bh / NH;
    const int h    = bh % NH;
    const int kvh  = h / GQA;
    const int q0   = qt * 64;
    const int tid  = threadIdx.x;
    const int wave = tid >> 5;
    const int lane = tid & 31;
    const int l16  = lane & 15;
    const int lh   = lane >> 4;
    const float scale = 0.08838834764831845f;           // 1/sqrt(128)

    // ---- load Q tile [64 x 128] via async copies ----
    for (int c = tid; c < 64 * HD / 8; c += 128) {
        int row = c >> 4, off = (c & 15) * 8;
        async_b128(lds_off32(lQ + row * HD + off),
                   q + (size_t)(b * SEQ + q0 + row) * QCOLS + h * HD + off);
    }
    wait_async0();
    __syncthreads();

    // preload this wave's Q A-fragments (rows = q0 + wave*16 + l16)
    v16bf qa[4];
#pragma unroll
    for (int ks = 0; ks < 4; ++ks) {
        const bf16_t* p = lQ + (wave * 16 + l16) * HD + ks * 32 + lh * 8;
        qa[ks] = ldfrag2(p, p + 16);
    }

    float m_row[8], l_row[8];
    v8f   oacc[8];
#pragma unroll
    for (int r = 0; r < 8; ++r) { m_row[r] = -1e30f; l_row[r] = 0.0f; }
#pragma unroll
    for (int j = 0; j < 8; ++j) oacc[j] = v8f_zero();

    for (int kb = 0; kb <= qt; ++kb) {
        const int k0 = kb * 64;
        __syncthreads();   // previous iteration done with lK/lV
        // ---- K tile [64 x 128] row-major via async copies ----
        for (int c = tid; c < 64 * HD / 8; c += 128) {
            int row = c >> 4, off = (c & 15) * 8;
            async_b128(lds_off32(lK + row * HD + off),
                       k + (size_t)(b * SEQ + k0 + row) * KVCOLS + kvh * HD + off);
        }
        // ---- V tile transposed (manual scatter): lV[d][key] ----
        for (int p = tid; p < 64 * (HD / 2); p += 128) {
            int key = p >> 6, dp = p & 63;
            union { unsigned u; bf16_t hx[2]; } uu;
            uu.u = *reinterpret_cast<const unsigned*>(
                v + (size_t)(b * SEQ + k0 + key) * KVCOLS + kvh * HD + dp * 2);
            lV[(2 * dp)     * 64 + key] = uu.hx[0];
            lV[(2 * dp + 1) * 64 + key] = uu.hx[1];
        }
        wait_async0();
        __syncthreads();

        // ---- S = scale * Q K^T  (wave: 16 rows x 64 keys = 4 n-tiles) ----
        v8f s[4];
#pragma unroll
        for (int nt = 0; nt < 4; ++nt) {
            s[nt] = v8f_zero();
#pragma unroll
            for (int ks = 0; ks < 4; ++ks) {
                const bf16_t* p = lK + (nt * 16 + l16) * HD + ks * 32 + lh * 8;
                s[nt] = wmma_bf16(qa[ks], ldfrag2(p, p + 16), s[nt]);
            }
        }
        // scale + causal mask
#pragma unroll
        for (int nt = 0; nt < 4; ++nt) {
            int col = k0 + nt * 16 + l16;
#pragma unroll
            for (int r = 0; r < 8; ++r) {
                float val = s[nt][r] * scale;
                int rowg = q0 + wave * 16 + r + 8 * lh;
                if (kb == qt && col > rowg) val = -1e30f;
                s[nt][r] = val;
            }
        }
        // ---- online softmax (row lives in a 16-lane half) ----
#pragma unroll
        for (int r = 0; r < 8; ++r) {
            float mx = -1e30f;
#pragma unroll
            for (int nt = 0; nt < 4; ++nt) mx = fmaxf(mx, s[nt][r]);
#pragma unroll
            for (int off = 1; off < 16; off <<= 1)
                mx = fmaxf(mx, __shfl_xor(mx, off, 16));
            float mnew  = fmaxf(m_row[r], mx);
            float alpha = __expf(m_row[r] - mnew);
            float rs = 0.0f;
#pragma unroll
            for (int nt = 0; nt < 4; ++nt) {
                float pv = __expf(s[nt][r] - mnew);
                s[nt][r] = pv;
                rs += pv;
            }
#pragma unroll
            for (int off = 1; off < 16; off <<= 1)
                rs += __shfl_xor(rs, off, 16);
            l_row[r] = l_row[r] * alpha + rs;
            m_row[r] = mnew;
#pragma unroll
            for (int j = 0; j < 8; ++j) oacc[j][r] *= alpha;
        }
        // ---- stage P (bf16) into per-wave LDS in row-major [16 x 64] ----
#pragma unroll
        for (int nt = 0; nt < 4; ++nt)
#pragma unroll
            for (int r = 0; r < 8; ++r)
                lP[wave][(r + 8 * lh) * 64 + nt * 16 + l16] = (bf16_t)s[nt][r];
        // same-wave LDS ops are in order: safe to read back immediately

        // ---- O += P V  (K-dim = 64 keys -> 2 k-steps, 8 n-tiles of dim) ----
#pragma unroll
        for (int ks2 = 0; ks2 < 2; ++ks2) {
            const bf16_t* pp = lP[wave] + l16 * 64 + ks2 * 32 + lh * 8;
            v16bf pa = ldfrag2(pp, pp + 16);
#pragma unroll
            for (int j = 0; j < 8; ++j) {
                const bf16_t* bp = lV + (j * 16 + l16) * 64 + ks2 * 32 + lh * 8;
                oacc[j] = wmma_bf16(pa, ldfrag2(bp, bp + 16), oacc[j]);
            }
        }
    }

    // ---- normalize + store bf16 [b,s,h,d] ----
#pragma unroll
    for (int j = 0; j < 8; ++j)
#pragma unroll
        for (int r = 0; r < 8; ++r) {
            float val = oacc[j][r] / l_row[r];
            int rowg = b * SEQ + q0 + wave * 16 + r + 8 * lh;
            int colg = h * HD + j * 16 + l16;
            o[(size_t)rowg * QCOLS + colg] = (bf16_t)val;
        }
}

// ---------------- host-side orchestration ----------------
extern "C" void kernel_launch(void* const* d_in, const int* in_sizes, int n_in,
                              void* d_out, int out_size, void* d_ws, size_t ws_size,
                              hipStream_t stream) {
    (void)in_sizes; (void)n_in; (void)out_size; (void)ws_size;
    const float* x  = (const float*)d_in[0];
    const float* fc = (const float*)d_in[1];
    const float* fs = (const float*)d_in[2];
    const float* wq = (const float*)d_in[3];
    const float* wk = (const float*)d_in[4];
    const float* wv = (const float*)d_in[5];
    const float* wo = (const float*)d_in[6];
    float* out = (float*)d_out;

    const long nx  = (long)MTOT * DIM;       // 16.7M
    const long nwq = (long)QCOLS * DIM;      // 16.7M
    const long nwk = (long)KVCOLS * DIM;     // 4.2M
    const long nq  = (long)MTOT * QCOLS;
    const long nk  = (long)MTOT * KVCOLS;

    char* w = (char*)d_ws;
    size_t off = 0;
    auto take = [&](long elems) { bf16_t* p = (bf16_t*)(w + off);
                                  off = (off + (size_t)elems * 2 + 255) & ~(size_t)255;
                                  return p; };
    bf16_t* xb  = take(nx);
    bf16_t* wqb = take(nwq);
    bf16_t* wkb = take(nwk);
    bf16_t* wvb = take(nwk);
    bf16_t* wob = take(nwq);
    bf16_t* qb  = take(nq);
    bf16_t* kb  = take(nk);
    bf16_t* vb  = take(nk);
    bf16_t* ob  = xb;   // attention output aliases x-bf16 (x dead after projections)

    // 1) fp32 -> bf16 conversions
    cvt_bf16_kernel<<<4096, 256, 0, stream>>>(x,  xb,  nx);
    cvt_bf16_kernel<<<4096, 256, 0, stream>>>(wq, wqb, nwq);
    cvt_bf16_kernel<<<2048, 256, 0, stream>>>(wk, wkb, nwk);
    cvt_bf16_kernel<<<2048, 256, 0, stream>>>(wv, wvb, nwk);
    cvt_bf16_kernel<<<4096, 256, 0, stream>>>(wo, wob, nwq);

    // 2) projections (bf16 WMMA, bf16 out)
    gemm_bf16_nt<false><<<dim3(QCOLS / 128,  MTOT / 128), 256, 0, stream>>>(
        xb, wqb, nullptr, qb, MTOT, QCOLS, DIM);
    gemm_bf16_nt<false><<<dim3(KVCOLS / 128, MTOT / 128), 256, 0, stream>>>(
        xb, wkb, nullptr, kb, MTOT, KVCOLS, DIM);
    gemm_bf16_nt<false><<<dim3(KVCOLS / 128, MTOT / 128), 256, 0, stream>>>(
        xb, wvb, nullptr, vb, MTOT, KVCOLS, DIM);

    // 3) RoPE on q and k
    rope_kernel<<<8192, 256, 0, stream>>>(qb, fc, fs, NH);
    rope_kernel<<<2048, 256, 0, stream>>>(kb, fc, fs, NKVH);

    // 4) causal GQA flash attention
    flash_attn_kernel<<<dim3(SEQ / 64, BATCH * NH), 128, 0, stream>>>(qb, kb, vb, ob);

    // 5) output projection (fp32 out)
    gemm_bf16_nt<true><<<dim3(DIM / 128, MTOT / 128), 256, 0, stream>>>(
        ob, wob, out, nullptr, MTOT, DIM, QCOLS);
}
